// ConvLSTM3D_87325275062605
// MI455X (gfx1250) — matrix operand order repaired and checked by reading, not compiled, not run on hardware
//
#include <hip/hip_runtime.h>
#include <hip/hip_bf16.h>
#include <math.h>

typedef __attribute__((ext_vector_type(16))) __bf16 v16bf;
typedef __attribute__((ext_vector_type(8)))  __bf16 v8bf;
typedef __attribute__((ext_vector_type(8)))  float  v8f;
typedef __attribute__((ext_vector_type(4)))  int    v4i;

#define B_      8
#define CIN     4
#define T_      8
#define DD      16
#define HH      32
#define WW      32
#define COUT    64
#define NCH     256              /* 4*C_OUT */
#define KIN     68               /* C_IN + C_OUT */
#define SPAT    (DD*HH*WW)       /* 16384 */
#define MTOT    (B_*SPAT)        /* 131072 */

#define MT   128
#define NT   64
#define NKCH 58                  /* 54 H-chunks + 4 X-chunks of K=32 */
#define LDSP 40                  /* padded LDS row stride (bf16 elems), 80B */

// ---- gfx1250 async-to-LDS path (guarded: falls back to sync vector copies) ----
#if defined(__AMDGCN__) && __has_builtin(__builtin_amdgcn_global_load_async_to_lds_b128)
#define ASYNC_LDS 1
#define ASYNC_B128(g, l)                                                        \
  __builtin_amdgcn_global_load_async_to_lds_b128(                               \
      (__attribute__((address_space(1))) v4i*)(g),                              \
      (__attribute__((address_space(3))) v4i*)(l), 0, 0)
#else
#define ASYNC_LDS 0
#endif

__device__ __forceinline__ void wait_async_lds() {
#if ASYNC_LDS
#if __has_builtin(__builtin_amdgcn_s_wait_asynccnt)
  __builtin_amdgcn_s_wait_asynccnt(0);
#else
  asm volatile("s_wait_asynccnt 0" ::: "memory");
#endif
#endif
}

__global__ __launch_bounds__(256) void zero_ws(float* p, int n) {
  int i = blockIdx.x * blockDim.x + threadIdx.x;
  if (i < n) p[i] = 0.f;
}

// Repack Wc fp32 [256][68][27] -> Wb bf16 [58 chunks][256][32]
//   chunk < 54 : tap = chunk/2, H-channel hc = (chunk&1)*32 + kk  (Wc index 4+hc)
//   chunk >= 54: xi = (chunk-54)*32+kk, tap = xi/4, X-channel c = xi&3 (pad: tap>=27 -> 0)
__global__ __launch_bounds__(256) void prep_weights(const float* __restrict__ Wc,
                                                    __bf16* __restrict__ Wb) {
  int i = blockIdx.x * blockDim.x + threadIdx.x;
  if (i >= NKCH * NCH * 32) return;
  int chunk = i / (NCH * 32);
  int n     = (i / 32) % NCH;
  int kk    = i & 31;
  float v = 0.f;
  if (chunk < 54) {
    int tap = chunk >> 1;
    int hc  = (chunk & 1) * 32 + kk;
    v = Wc[(n * KIN + 4 + hc) * 27 + tap];
  } else {
    int xi  = (chunk - 54) * 32 + kk;
    int tap = xi >> 2;
    int c   = xi & 3;
    if (tap < 27) v = Wc[(n * KIN + c) * 27 + tap];
  }
  Wb[i] = (__bf16)v;
}

// ---- helpers for the GEMM mainloop ----
__device__ __forceinline__ void store_a_row(__bf16* dst, const float* vals) {
  #pragma unroll
  for (int g = 0; g < 4; ++g) {
    v8bf pk;
    #pragma unroll
    for (int j = 0; j < 8; ++j) pk[j] = (__bf16)vals[g * 8 + j];
    *(v8bf*)(dst + g * 8) = pk;
  }
}

__device__ __forceinline__ void fill_b(const __bf16* __restrict__ Wb, __bf16* Bs,
                                       int chunk, int n0, int tid) {
  const int n  = tid >> 1;
  const int ho = (tid & 1) * 16;
  const __bf16* src = &Wb[((size_t)chunk * NCH + n0 + n) * 32 + ho];
  __bf16* dst = &Bs[n * LDSP + ho];
#if ASYNC_LDS
  ASYNC_B128(src, dst);
  ASYNC_B128(src + 8, dst + 8);
#else
  *(v8bf*)dst       = *(const v8bf*)src;
  *(v8bf*)(dst + 8) = *(const v8bf*)(src + 8);
#endif
  if (chunk + 1 < NKCH)
    __builtin_prefetch(&Wb[((size_t)(chunk + 1) * NCH + n0 + n) * 32], 0, 1);
}

__device__ __forceinline__ void mma_step(const __bf16* As, const __bf16* Bs,
                                         v8f acc[8], int arow0, int ab0, int bo0,
                                         int bn) {
  v16bf afr[2];
  #pragma unroll
  for (int am = 0; am < 2; ++am) {
    const __bf16* ap = &As[(arow0 + am * 16) * LDSP];
    v8bf lo = *(const v8bf*)(ap + ab0);
    v8bf hi = *(const v8bf*)(ap + ab0 + 16);
    #pragma unroll
    for (int e = 0; e < 8; ++e) { afr[am][e] = lo[e]; afr[am][8 + e] = hi[e]; }
  }
  #pragma unroll
  for (int f = 0; f < 4; ++f) {
    const __bf16* bp = &Bs[(f * 16 + bn) * LDSP + bo0];
    v8bf blo = *(const v8bf*)(bp);
    v8bf bhi = *(const v8bf*)(bp + 16);
    v16bf bfr;
    #pragma unroll
    for (int e = 0; e < 8; ++e) { bfr[e] = blo[e]; bfr[8 + e] = bhi[e]; }
    acc[f * 2 + 0] = __builtin_amdgcn_wmma_f32_16x16x32_bf16(
        false, afr[0], false, bfr, (short)0, acc[f * 2 + 0], false, false);
    acc[f * 2 + 1] = __builtin_amdgcn_wmma_f32_16x16x32_bf16(
        false, afr[1], false, bfr, (short)0, acc[f * 2 + 1], false, false);
  }
}

// Implicit-GEMM conv over K = 27 taps x 68 channels, bf16 WMMA, f32 accum.
__global__ __launch_bounds__(128) void conv_gemm(const float* __restrict__ X,
                                                 const float* __restrict__ Hst,
                                                 const __bf16* __restrict__ Wb,
                                                 float* __restrict__ conv, int t) {
  __shared__ __bf16 As[MT * LDSP];
  __shared__ __bf16 Bs[NT * LDSP];
  const int tid  = threadIdx.x;
  const int lane = tid & 31;
  const int wv   = tid >> 5;            // wave 0..3, owns 32 M rows x 64 N
  const int m0   = blockIdx.x * MT;
  const int n0   = blockIdx.y * NT;

  // per-thread A row: one LDS row per thread, constant across all k-steps
  const int bbm = m0 >> 14;             // batch is block-constant (128 | 16384)
  const int s   = (m0 + tid) & (SPAT - 1);
  const int d   = s >> 10;
  const int h   = (s >> 5) & 31;
  const int w   = s & 31;

  // constant fragment addressing (documented 16-bit WMMA lane layouts)
  const int arow0 = wv * 32 + (lane & 15);
  const int ab0   = (lane < 16) ? 0 : 8;     // A: K 0..7 / 8..15, second half +16
  const int bo0   = (lane < 16) ? 0 : 8;     // B: K 0..7 / 8..15, second half +16
  const int bn    = lane & 15;

  v8f acc[8] = {};
  float vals[32];
  __bf16* arow = &As[tid * LDSP];
  const float* hbase = Hst + ((size_t)(bbm * COUT)) * SPAT;

  // ---------------- H-state part: 27 taps x 64 channels ----------------
  int chunk = 0;
  for (int kd = -1; kd <= 1; ++kd) {
    const int dz = d + kd;
    for (int kh = -1; kh <= 1; ++kh) {
      const int hz = h + kh;
      for (int kw = -1; kw <= 1; ++kw) {
        const int wz = w + kw;
        const bool inb = (unsigned)dz < (unsigned)DD && (unsigned)hz < (unsigned)HH &&
                         (unsigned)wz < (unsigned)WW;
        const int szoff = (dz * HH + hz) * WW + wz;

        #pragma unroll
        for (int half = 0; half < 2; ++half, ++chunk) {
          if (inb) {
            const float* p = hbase + (size_t)(half * 32) * SPAT + szoff;
            #pragma unroll
            for (int i = 0; i < 32; ++i) vals[i] = p[(size_t)i * SPAT];
          } else {
            #pragma unroll
            for (int i = 0; i < 32; ++i) vals[i] = 0.f;
          }
          store_a_row(arow, vals);
          fill_b(Wb, Bs, chunk, n0, tid);
          wait_async_lds();
          __syncthreads();
          mma_step(As, Bs, acc, arow0, ab0, bo0, bn);
          __syncthreads();
        }
      }
    }
  }

  // ---------------- X part: 4 chunks of (8 taps x 4 channels), fully unrolled ----
  const float* xbase = X + ((size_t)(bbm * CIN) * T_ + t) * SPAT;
  #pragma unroll
  for (int xc = 0; xc < 4; ++xc) {
    const int ch = 54 + xc;
    #pragma unroll
    for (int tp = 0; tp < 8; ++tp) {
      const int tap = xc * 8 + tp;              // constant after unroll
      float v0 = 0.f, v1 = 0.f, v2 = 0.f, v3 = 0.f;
      if (tap < 27) {
        const int kd = tap / 9 - 1;             // folds to constants
        const int kh = (tap / 3) % 3 - 1;
        const int kw = tap % 3 - 1;
        const int dz = d + kd, hz = h + kh, wz = w + kw;
        if ((unsigned)dz < (unsigned)DD && (unsigned)hz < (unsigned)HH &&
            (unsigned)wz < (unsigned)WW) {
          const float* p = xbase + (dz * HH + hz) * WW + wz;
          v0 = p[0];
          v1 = p[(size_t)T_ * SPAT];
          v2 = p[(size_t)2 * T_ * SPAT];
          v3 = p[(size_t)3 * T_ * SPAT];
        }
      }
      vals[tp * 4 + 0] = v0; vals[tp * 4 + 1] = v1;
      vals[tp * 4 + 2] = v2; vals[tp * 4 + 3] = v3;
    }
    store_a_row(arow, vals);
    fill_b(Wb, Bs, ch, n0, tid);
    wait_async_lds();
    __syncthreads();
    mma_step(As, Bs, acc, arow0, ab0, bo0, bn);
    __syncthreads();
  }

  // ---- epilogue: consecutive r -> consecutive addresses (merges to b128) ----
  const int sbase = (m0 & (SPAT - 1)) + wv * 32 + ((lane >> 4) << 3);
  float* cbase = conv + ((size_t)(bbm * NCH)) * SPAT + sbase;
  #pragma unroll
  for (int f = 0; f < 4; ++f) {
    float* cp = cbase + (size_t)(n0 + f * 16 + bn) * SPAT;
    #pragma unroll
    for (int am = 0; am < 2; ++am) {
      #pragma unroll
      for (int r = 0; r < 8; ++r) cp[am * 16 + r] = acc[f * 2 + am][r];
    }
  }
}

__device__ __forceinline__ float sigmoidf_(float x) { return 1.f / (1.f + __expf(-x)); }

__global__ __launch_bounds__(256) void gate_update(
    const float* __restrict__ conv, const float* __restrict__ bias,
    const float* __restrict__ Wci, const float* __restrict__ Wcf,
    const float* __restrict__ Wco, float* __restrict__ Hst,
    float* __restrict__ Cst, float* __restrict__ out, int t) {
  int idx = blockIdx.x * blockDim.x + threadIdx.x;   // (bb*COUT+co)*SPAT + s
  if (idx >= B_ * COUT * SPAT) return;
  int s  = idx & (SPAT - 1);
  int r  = idx >> 14;
  int co = r & (COUT - 1);
  int bb = r >> 6;
  size_t cb = ((size_t)(bb * NCH)) * SPAT + s;
  float ic = conv[cb + (size_t)co * SPAT]              + bias[co];
  float fc = conv[cb + (size_t)(co + COUT) * SPAT]     + bias[co + COUT];
  float cc = conv[cb + (size_t)(co + 2 * COUT) * SPAT] + bias[co + 2 * COUT];
  float oc = conv[cb + (size_t)(co + 3 * COUT) * SPAT] + bias[co + 3 * COUT];
  float Cs = Cst[idx];
  int ps = co * SPAT + s;
  float ig = sigmoidf_(ic + Wci[ps] * Cs);
  float fg = sigmoidf_(fc + Wcf[ps] * Cs);
  float Cn = fg * Cs + ig * tanhf(cc);
  float og = sigmoidf_(oc + Wco[ps] * Cn);
  float Hn = og * tanhf(Cn);
  Cst[idx] = Cn;
  Hst[idx] = Hn;
  out[(((size_t)(bb * COUT + co)) * T_ + t) * SPAT + s] = Hn;
}

extern "C" void kernel_launch(void* const* d_in, const int* in_sizes, int n_in,
                              void* d_out, int out_size, void* d_ws, size_t ws_size,
                              hipStream_t stream) {
  const float* X    = (const float*)d_in[0];
  const float* Wc   = (const float*)d_in[1];
  const float* bias = (const float*)d_in[2];
  const float* Wci  = (const float*)d_in[3];
  const float* Wcf  = (const float*)d_in[4];
  const float* Wco  = (const float*)d_in[5];
  float* out = (float*)d_out;

  char* ws = (char*)d_ws;
  float*  Hst  = (float*)(ws);                      // 33.5 MB
  float*  Cst  = (float*)(ws + 33554432);           // 33.5 MB
  float*  conv = (float*)(ws + 67108864);           // 134 MB
  __bf16* Wb   = (__bf16*)(ws + 201326592);         // 0.95 MB

  // zero H and C states (contiguous)
  int nstate = B_ * COUT * SPAT * 2;
  zero_ws<<<(nstate + 255) / 256, 256, 0, stream>>>(Hst, nstate);

  // repack weights to bf16 unified-K layout [58][256][32]
  int nw = NKCH * NCH * 32;
  prep_weights<<<(nw + 255) / 256, 256, 0, stream>>>(Wc, Wb);

  dim3 grid(MTOT / MT, NCH / NT);   // 1024 x 4 blocks, 128 threads (4 waves)
  int ngate = B_ * COUT * SPAT;
  for (int t = 0; t < T_; ++t) {
    conv_gemm<<<grid, 128, 0, stream>>>(X, Hst, Wb, conv, t);
    gate_update<<<(ngate + 255) / 256, 256, 0, stream>>>(conv, bias, Wci, Wcf, Wco,
                                                         Hst, Cst, out, t);
  }
}
